// VCodec_4604204942039
// MI455X (gfx1250) — matrix operand order repaired and checked by reading, not compiled, not run on hardware
//
#include <hip/hip_runtime.h>
#include <hip/hip_bf16.h>
#include <stddef.h>

typedef __attribute__((ext_vector_type(16))) _Float16 v16h;
typedef __attribute__((ext_vector_type(8)))  float    v8f;

// Problem constants (from reference)
#define BQ   16
#define TQ   2048
#define BT   (BQ * TQ)     // 32768 rows
#define DQ   100
#define GN   2
#define DG   50            // per-group dims (= codebook dim)
#define QN   4             // quantizers per group
#define KN   1024          // codebook entries

// Output layout (floats): quantized | loss | commit | recon
#define OUT_Q    (BT * DQ)          // 3,276,800
#define OUT_LOSS (OUT_Q)            // + BT
#define OUT_CL   (OUT_Q + BT)       // + 1
#define OUT_REC  (OUT_Q + BT + 1)   // + BT

// Workspace layout (float offsets)
#define WS_CNORM  0                 // G*Q*K = 8192 floats
#define WS_RECON  8192              // BT floats
#define WS_COMMIT 40960             // 1 float
#define WS_PCB    40976             // packed f16 codebooks start (32B aligned)
// packed codebook: [gq(8)][nt(64)][frag(2)][lane(32)] x v16h = 524288 f16 = 1MB
#define PCB_V16H_COUNT (8 * 64 * 2 * 32)
#define WS_NEED_BYTES  ((size_t)WS_PCB * 4 + (size_t)PCB_V16H_COUNT * 32)

static __device__ __forceinline__ v8f wmma_f16(v16h a, v16h b, v8f c) {
  return __builtin_amdgcn_wmma_f32_16x16x32_f16(
      /*neg_a=*/false, a, /*neg_b=*/false, b,
      /*c_mod=*/(short)0, c, /*reuse_a=*/false, /*reuse_b=*/false);
}

// ---------------- zero accumulators ----------------
__global__ void rvq_zero_kernel(float* __restrict__ ws) {
  int tid = blockIdx.x * blockDim.x + threadIdx.x;
  if (tid < BT) ws[WS_RECON + tid] = 0.0f;
  if (tid == 0) ws[WS_COMMIT] = 0.0f;
}

// ---------------- ||cb||^2 precompute ----------------
__global__ void rvq_cnorm_kernel(const float* __restrict__ codebooks,
                                 float* __restrict__ ws) {
  int tid = blockIdx.x * blockDim.x + threadIdx.x;  // gq*K + k
  if (tid >= GN * QN * KN) return;
  const float* p = codebooks + (size_t)tid * DG;
  float s = 0.0f;
#pragma unroll
  for (int i = 0; i < DG; ++i) s += p[i] * p[i];
  ws[WS_CNORM + tid] = s;
}

// ---------------- pack codebooks into f16 B-fragment layout ----------------
// B fragment (16-bit, 32x16, 8 VGPRs as v16h per lane):
//   lanes 0-15  : column N = lane,    K = frag*32 + {0..7,16..23}
//   lanes 16-31 : column N = lane-16, K = frag*32 + {8..15,24..31}
__global__ void rvq_pack_kernel(const float* __restrict__ codebooks,
                                float* __restrict__ ws) {
  int tid = blockIdx.x * blockDim.x + threadIdx.x;  // [0, 8*64*2*32)
  if (tid >= PCB_V16H_COUNT) return;
  int lane = tid & 31;
  int frag = (tid >> 5) & 1;
  int nt   = (tid >> 6) & 63;
  int gq   = tid >> 12;
  int code   = nt * 16 + (lane & 15);
  int kstart = frag * 32 + (lane >> 4) * 8;
  const float* src = codebooks + ((size_t)gq * KN + code) * DG;
  v16h tmp;
#pragma unroll
  for (int i = 0; i < 16; ++i) {
    int k = kstart + (i < 8 ? i : i + 8);
    float v = (k < DG) ? src[k] : 0.0f;
    tmp[i] = (_Float16)v;
  }
  ((v16h*)(ws + WS_PCB))[tid] = tmp;
}

// ---------------- fused RVQ kernel ----------------
// Block: 128 threads (4 waves). Each block: 64 rows of one group.
// grid.x = GN * (BT/64) = 1024. PACKED chosen at compile time -> branch-free
// inner loop, software-pipelined B-fragment loads.
template <bool PACKED>
__global__ __launch_bounds__(128)
void rvq_fused_kernel(const float* __restrict__ x,
                      const float* __restrict__ W_in,
                      const float* __restrict__ b_in,
                      const float* __restrict__ W_out,
                      const float* __restrict__ b_out,
                      const float* __restrict__ codebooks,
                      float* __restrict__ out,
                      float* __restrict__ ws) {
  __shared__ float xs[64][52];   // x tile (group columns)
  __shared__ float rl[64][52];   // residual r
  __shared__ float ql[64][52];   // accumulated quantized output
  __shared__ float wl[DG * DG];  // W_in then W_out
  __shared__ int   idxRow[64];

  const int tid  = threadIdx.x;
  const int lane = tid & 31;
  const int wv   = tid >> 5;          // wave id 0..3 -> M-tile
  const int g    = blockIdx.x & 1;
  const int m0   = (blockIdx.x >> 1) * 64;

  const int row = tid >> 1;           // 0..63
  const int c0  = (tid & 1) * 25;     // column half

  // ---- stage x tile, zero qout, stage W_in ----
#pragma unroll
  for (int i = 0; i < 25; ++i) {
    int e = tid * 25 + i;
    int rr = e / 50, cc = e % 50;
    float xv = x[(size_t)(m0 + rr) * DQ + g * DG + cc];
    xs[rr][cc] = xv;
    ql[rr][cc] = 0.0f;
  }
  for (int i = tid; i < DG * DG; i += 128) wl[i] = W_in[g * DG * DG + i];
  __syncthreads();

  // ---- project_in: r = x @ Wi + bi (f32) ----
  for (int i = 0; i < 25; ++i) {
    int c = c0 + i;
    float acc = b_in[g * DG + c];
#pragma unroll
    for (int d = 0; d < DG; ++d) acc += xs[row][d] * wl[d * DG + c];
    rl[row][c] = acc;
  }

  const int mrow = wv * 16 + (lane & 15);   // local row of this lane's A fragment
  const int kb   = (lane >> 4) * 8;         // A-layout K sub-base (0 or 8)
  const v16h* pcbv = (const v16h*)(ws + WS_PCB);
  float commitAcc = 0.0f;

  for (int q = 0; q < QN; ++q) {
    __syncthreads();  // r ready for everyone
    const int gq = g * QN + q;
    const float* cbs = codebooks + (size_t)gq * KN * DG;
    const float* cn  = ws + WS_CNORM + gq * KN;

    // ---- build A fragments from r (f32 LDS -> f16), K padded to 64 ----
    // 16-bit A 16x32 layout: lanes<16: K {0..7,16..23}; lanes>=16: K {8..15,24..31}
    v16h a0, a1;
#pragma unroll
    for (int i = 0; i < 8; ++i) {
      a0[i]     = (_Float16)rl[mrow][kb + i];
      a0[8 + i] = (_Float16)rl[mrow][16 + kb + i];
      a1[i]     = (_Float16)rl[mrow][32 + kb + i];      // 32..47 < 50: valid
      int k3 = 48 + kb + i;                              // 48..63: guard
      float v = rl[mrow][(k3 < DG) ? k3 : 0];
      a1[8 + i] = (_Float16)((k3 < DG) ? v : 0.0f);
    }

    float bestd[8];
    int   bestn[8];
#pragma unroll
    for (int j = 0; j < 8; ++j) { bestd[j] = 3.0e38f; bestn[j] = 0; }

    if (PACKED) {
      // Per-lane base pointers; stride between nt tiles is 64 v16h.
      const v16h*  pl  = pcbv + (size_t)gq * 64 * 2 * 32 + lane;
      const float* cnl = cn + (lane & 15);

      // Software pipeline: preload tile 0, issue tile (nt+1)&63 inside the
      // body (wrap avoids a guard branch; the wrapped load is valid memory),
      // compute WMMA on current registers. In-order LOADcnt puts the wait a
      // full iteration after issue, hiding L2 latency.
      v16h  b0  = pl[0];
      v16h  b1  = pl[32];
      float cnv = cnl[0];
#pragma unroll 2
      for (int nt = 0; nt < 64; ++nt) {
        const size_t nx = (size_t)((nt + 1) & 63);
        v16h  nb0 = pl[nx * 64];
        v16h  nb1 = pl[nx * 64 + 32];
        float ncn = cnl[nx * 16];
        __builtin_prefetch((const void*)&pl[(size_t)((nt + 2) & 63) * 64], 0, 3);

        v8f c = {0.0f, 0.0f, 0.0f, 0.0f, 0.0f, 0.0f, 0.0f, 0.0f};
        c = wmma_f16(a0, b0, c);
        c = wmma_f16(a1, b1, c);

        const int ncode = nt * 16 + (lane & 15);
#pragma unroll
        for (int j = 0; j < 8; ++j) {
          float dist = cnv - 2.0f * c[j];   // ||cb||^2 - 2 r.cb
          bool p = dist < bestd[j];
          bestd[j] = p ? dist : bestd[j];
          bestn[j] = p ? ncode : bestn[j];
        }
        b0 = nb0; b1 = nb1; cnv = ncn;
      }
    } else {
      for (int nt = 0; nt < 64; ++nt) {
        int code  = nt * 16 + (lane & 15);
        int khalf = (lane >> 4) * 8;
        const float* cr = cbs + (size_t)code * DG;
        v16h b0, b1;
#pragma unroll
        for (int i = 0; i < 16; ++i) {
          b0[i] = (_Float16)cr[khalf + (i < 8 ? i : i + 8)];               // K 0..31 valid
          int k = 32 + khalf + (i < 8 ? i : i + 8);
          float v = (k < DG) ? cr[k] : 0.0f;
          b1[i] = (_Float16)v;
        }
        v8f c = {0.0f, 0.0f, 0.0f, 0.0f, 0.0f, 0.0f, 0.0f, 0.0f};
        c = wmma_f16(a0, b0, c);
        c = wmma_f16(a1, b1, c);
        float cnv = cn[code];
#pragma unroll
        for (int j = 0; j < 8; ++j) {
          float dist = cnv - 2.0f * c[j];
          bool p = dist < bestd[j];
          bestd[j] = p ? dist : bestd[j];
          bestn[j] = p ? code : bestn[j];
        }
      }
    }

    // ---- argmin across N: reduce over the 16 lanes holding each row ----
    // C layout: element (M,N) at vgpr j, lane l: M = j + 8*(l>=16), N = l%16
    // Branch-free ternary selects -> v_cndmask, no EXEC manipulation.
#pragma unroll
    for (int j = 0; j < 8; ++j) {
      float v = bestd[j];
      int   n = bestn[j];
#pragma unroll
      for (int off = 8; off > 0; off >>= 1) {
        float ov = __shfl_xor(v, off, 16);
        int   on = __shfl_xor(n, off, 16);
        const bool p = (ov < v) || ((ov == v) && (on < n));
        v = p ? ov : v;
        n = p ? on : n;
      }
      if ((lane & 15) == 0) idxRow[wv * 16 + j + 8 * (lane >> 4)] = n;
    }
    __syncthreads();

    // ---- gather + straight-through update (f32, matches reference) ----
    {
      int idx = idxRow[row];
      const float* cbrow = cbs + (size_t)idx * DG;
#pragma unroll
      for (int i = 0; i < 25; ++i) {
        int c = c0 + i;
        float qv = cbrow[c];
        float rr = rl[row][c];
        float df = qv - rr;                 // stop_grad(qv - r)
        commitAcc += df * df;
        ql[row][c] += rr + df;              // qout += r + (qv - r)
        rl[row][c] = rr - qv;               // r -= qv
      }
    }
  }
  __syncthreads();

  // ---- project_out: out = qout @ Wo + bo; quantized store + recon ----
  for (int i = tid; i < DG * DG; i += 128) wl[i] = W_out[g * DG * DG + i];
  __syncthreads();

  float rpart = 0.0f;
  const size_t grow = (size_t)(m0 + row);
  for (int i = 0; i < 25; ++i) {
    int c = c0 + i;
    float acc = b_out[g * DG + c];
#pragma unroll
    for (int d = 0; d < DG; ++d) acc += ql[row][d] * wl[d * DG + c];
    out[grow * DQ + g * DG + c] = acc;
    float df = xs[row][c] - acc;
    rpart += df * df;
  }
  atomicAdd(&ws[WS_RECON + grow], rpart);
  atomicAdd(&ws[WS_COMMIT], commitAcc);
}

// ---------------- finalize losses ----------------
__global__ void rvq_final_kernel(float* __restrict__ out,
                                 const float* __restrict__ ws) {
  int tid = blockIdx.x * blockDim.x + threadIdx.x;
  if (tid >= BT) return;
  float cl  = ws[WS_COMMIT] * (1.0f / ((float)GN * QN * BT * DG));
  float rec = ws[WS_RECON + tid] * (1.0f / DQ);
  out[OUT_LOSS + tid] = rec + cl;
  out[OUT_REC + tid]  = rec;
  if (tid == 0) out[OUT_CL] = cl;
}

extern "C" void kernel_launch(void* const* d_in, const int* in_sizes, int n_in,
                              void* d_out, int out_size, void* d_ws, size_t ws_size,
                              hipStream_t stream) {
  const float* x         = (const float*)d_in[0];
  const float* W_in      = (const float*)d_in[1];
  const float* b_in      = (const float*)d_in[2];
  const float* W_out     = (const float*)d_in[3];
  const float* b_out     = (const float*)d_in[4];
  const float* codebooks = (const float*)d_in[5];
  float* out = (float*)d_out;
  float* ws  = (float*)d_ws;

  const bool packed = (ws_size >= WS_NEED_BYTES);

  rvq_zero_kernel<<<(BT + 255) / 256, 256, 0, stream>>>(ws);
  rvq_cnorm_kernel<<<(GN * QN * KN + 255) / 256, 256, 0, stream>>>(codebooks, ws);
  if (packed) {
    rvq_pack_kernel<<<PCB_V16H_COUNT / 256, 256, 0, stream>>>(codebooks, ws);
    rvq_fused_kernel<true><<<GN * (BT / 64), 128, 0, stream>>>(
        x, W_in, b_in, W_out, b_out, codebooks, out, ws);
  } else {
    rvq_fused_kernel<false><<<GN * (BT / 64), 128, 0, stream>>>(
        x, W_in, b_in, W_out, b_out, codebooks, out, ws);
  }
  rvq_final_kernel<<<(BT + 255) / 256, 256, 0, stream>>>(out, ws);
}
